// CorefDecoderHOI_3444563771557
// MI455X (gfx1250) — matrix operand order repaired and checked by reading, not compiled, not run on hardware
//
#include <hip/hip_runtime.h>

// ---------------- problem constants (from reference) ----------------
#define NSPANS 2048
#define MAXA   50
#define APAD   64            // 50 antecedents padded to 4 M-tiles of 16
#define DSPAN  2324          // SPAN_DIM
#define PAIRK  6992          // 3*SPAN_DIM + FEAT
#define KPAD   7168          // PAIRK padded to multiple of 256 (zero rows)
#define UN     512           // UNARY
#define FEATN  20
#define KSTAGE 256           // K columns staged in LDS per iteration
#define CHUNKS 8             // 8 WMMA K-chunks (32 each) per stage
#define NSTAGES (KPAD / KSTAGE)   // 28
#define CROWS  2049          // cemb rows (cluster ids can reach 2048)
#define SMAX   2052          // LDS state array size (padded)

typedef __attribute__((ext_vector_type(16))) __bf16 v16bf;
typedef __attribute__((ext_vector_type(8)))  float  v8f;
typedef __attribute__((ext_vector_type(4)))  unsigned int u32x4;

// -------- one-time W1 fp32 -> bf16 pack into WMMA B-fragment order -------
// B tile (kc, nt) holds K=kc*32..+31, N=nt*16..+15.  This code's contiguous-K B
// layout: lanes 0-15 hold K 0-15, lanes 16-31 hold K 16-31, 2 halves/VGPR in
// ascending K.  We store each lane's 16 halves contiguously -> runtime load
// is 2x b128 per fragment.
__global__ __launch_bounds__(256)
void pack_w1_bf16(const float* __restrict__ W1, __bf16* __restrict__ Bpack) {
  int idx = blockIdx.x * 256 + threadIdx.x;
  if (idx >= KPAD * UN) return;
  int tile = idx >> 9;          // 512 halves per tile
  int inT  = idx & 511;
  int lane = inT >> 4;
  int h    = inT & 15;
  int kc = tile >> 5;           // K-chunk
  int nt = tile & 31;           // N-tile
  int k = kc * 32 + (lane >> 4) * 16 + h;
  int n = nt * 16 + (lane & 15);
  float v = (k < PAIRK) ? W1[k * UN + n] : 0.0f;
  Bpack[idx] = (__bf16)v;
}

__global__ __launch_bounds__(256)
void zero_f32(float* __restrict__ p, int n) {
  int idx = blockIdx.x * 256 + threadIdx.x;
  if (idx < n) p[idx] = 0.0f;
}

// -------------------- persistent sequential decoder --------------------
__global__ __launch_bounds__(1024, 1)
void coref_main(const float* __restrict__ spanEmb,     // [2048,2324]
                const float* __restrict__ antScores,   // [2048,50]
                const float* __restrict__ clusSizeEmb, // [10,20]
                const float* __restrict__ b1,          // [512]
                const float* __restrict__ W2,          // [512]
                const float* __restrict__ b2,          // [1]
                const int*   __restrict__ antIdx,      // [2048,50]
                const __bf16* __restrict__ Bpack,      // packed W1
                float* __restrict__ cemb,              // [2049,2324] ws
                float* __restrict__ out) {             // [2048,50]
  __shared__ __align__(16) __bf16 pairLDS[APAD][KSTAGE];   // 32 KB
  __shared__ int   s2c[SMAX];
  __shared__ int   csz[SMAX];
  __shared__ float b1s[UN], w2s[UN];
  __shared__ float scorePart[32][16];
  __shared__ float scoreFin[APAD];
  __shared__ int   ls_acid[APAD];
  __shared__ int   ls_bucket[APAD];
  __shared__ int   decDo, decCid, decMa, decInit, nclusS;
  __shared__ float decSize;

  const int tid  = threadIdx.x;
  const int lane = tid & 31;
  const int wave = tid >> 5;
  const int lg   = lane >> 4;          // lane group (0/1)
  const int mt   = wave & 3;           // M-tile 0..3
  const int ntBase = (wave >> 2) * 4;  // 4 N-tiles per wave, covers 0..31

  for (int j = tid; j < SMAX; j += 1024) { s2c[j] = 0; csz[j] = 1; }
  for (int j = tid; j < UN;   j += 1024) { b1s[j] = b1[j]; w2s[j] = W2[j]; }
  if (tid == 0) nclusS = 1;
  __syncthreads();

  const v8f vzero = {0.f, 0.f, 0.f, 0.f, 0.f, 0.f, 0.f, 0.f};

  for (int i = 0; i < NSPANS; ++i) {
    const float* span = spanEmb + (size_t)i * DSPAN;

    // ---- gather antecedent cluster ids + size buckets ----
    if (tid < APAD) {
      int acid = 0, bucket = 0;
      if (tid < MAXA) {
        int idx = antIdx[i * MAXA + tid];
        acid = s2c[idx];
        int d = csz[acid];                       // >= 1 always
        if (d <= 4) bucket = d;
        else { bucket = (31 - __builtin_clz(d)) + 3; if (bucket > 9) bucket = 9; }
      }
      ls_acid[tid] = acid;
      ls_bucket[tid] = bucket;
    }
    __syncthreads();

    // ---- pair[64, KPAD] @ W1bf[KPAD, 512] via WMMA bf16 ----
    v8f acc[4];
#pragma unroll
    for (int t = 0; t < 4; ++t) acc[t] = vzero;

    const int mrow = mt * 16 + (lane & 15);      // A-matrix row for this lane

    for (int st = 0; st < NSTAGES; ++st) {
      const int kBase = st * KSTAGE;
      // build the pair tile (on-the-fly bf16 conversion)
      for (int e = tid; e < APAD * KSTAGE; e += 1024) {
        int m  = e >> 8;
        int kk = e & (KSTAGE - 1);
        int k  = kBase + kk;
        float v = 0.0f;
        if (m < MAXA) {
          int acid = ls_acid[m];
          if (k < DSPAN)            v = span[k];
          else if (k < 2 * DSPAN)   v = cemb[(size_t)acid * DSPAN + (k - DSPAN)];
          else if (k < 3 * DSPAN) { int kd = k - 2 * DSPAN;
                                    v = span[kd] * cemb[(size_t)acid * DSPAN + kd]; }
          else if (k < PAIRK)       v = clusSizeEmb[ls_bucket[m] * FEATN + (k - 3 * DSPAN)];
        }
        pairLDS[m][kk] = (__bf16)v;
      }
      __syncthreads();

      // prefetch next stage's B region (global_prefetch_b8)
      if (st + 1 < NSTAGES) {
        const __bf16* nb = Bpack + ((size_t)((st + 1) * CHUNKS) * 32 + ntBase) * 512;
        __builtin_prefetch(nb + (size_t)lane * 64, 0, 0);
      }

      for (int kc = 0; kc < CHUNKS; ++kc) {
        // A fragment: halves 0-15 = K (16*lg .. 16*lg+15), contiguous
        union { u32x4 u4[2]; v16bf v; } ua;
        const int kloc = kc * 32 + lg * 16;
        ua.u4[0] = *(const u32x4*)&pairLDS[mrow][kloc];
        ua.u4[1] = *(const u32x4*)&pairLDS[mrow][kloc + 8];
        const v16bf aF = ua.v;

        const int kcg = st * CHUNKS + kc;
        const __bf16* bbase = Bpack + ((size_t)kcg * 32) * 512 + (size_t)lane * 16;
#pragma unroll
        for (int t = 0; t < 4; ++t) {
          const __bf16* bp = bbase + (size_t)(ntBase + t) * 512;
          union { u32x4 u4[2]; v16bf v; } ub;
          ub.u4[0] = *(const u32x4*)(bp);
          ub.u4[1] = *(const u32x4*)(bp + 8);
          acc[t] = __builtin_amdgcn_wmma_f32_16x16x32_bf16(
              false, aF, false, ub.v, (short)0, acc[t], false, false);
        }
      }
      __syncthreads();
    }

    // ---- relu + W2 dot, deterministic reduction ----
    // C layout: n = lane&15, m = v + 8*lg  (per tile)
    float partial[8];
#pragma unroll
    for (int v = 0; v < 8; ++v) partial[v] = 0.0f;
#pragma unroll
    for (int t = 0; t < 4; ++t) {
      int u = (ntBase + t) * 16 + (lane & 15);
      float bb = b1s[u], w2v = w2s[u];
#pragma unroll
      for (int v = 0; v < 8; ++v) {
        float h = acc[t][v] + bb;
        h = h > 0.0f ? h : 0.0f;
        partial[v] += h * w2v;
      }
    }
#pragma unroll
    for (int v = 0; v < 8; ++v) {
      float p = partial[v];
      p += __shfl_xor(p, 1);
      p += __shfl_xor(p, 2);
      p += __shfl_xor(p, 4);
      p += __shfl_xor(p, 8);
      if ((lane & 15) == 0) scorePart[wave][(lg << 3) + v] = p;
    }
    __syncthreads();

    if (tid < APAD) {
      float s = 0.0f;
      int mtA = tid >> 4, ml = tid & 15;
#pragma unroll
      for (int j = 0; j < 8; ++j) s += scorePart[mtA + 4 * j][ml];
      float sc = (tid < MAXA && ls_acid[tid] > 0) ? (s + b2[0]) : 0.0f;
      scoreFin[tid] = sc;
      if (tid < MAXA) out[i * MAXA + tid] = sc;
    }
    __syncthreads();

    // ---- sequential decision (single thread, matches scan semantics) ----
    if (tid == 0) {
      float best = -3.4e38f; int mi = 0;
      for (int a = 0; a < MAXA; ++a) {
        float tot = antScores[i * MAXA + a] + scoreFin[a];
        if (tot > best) { best = tot; mi = a; }   // first-max like jnp.argmax
      }
      int doF = best >= 0.0f;
      int ma  = antIdx[i * MAXA + mi];
      int acid = s2c[ma];
      int isNew = (acid == 0);
      int cid = isNew ? nclusS : acid;
      if (cid > CROWS - 1) cid = CROWS - 1;
      int ini = doF && isNew;
      decDo = doF; decCid = cid; decMa = ma; decInit = ini;
      decSize = (float)csz[cid];                  // size before increment
      if (ini) { s2c[ma] = cid; nclusS++; }
      if (doF) { s2c[i] = cid; csz[cid] += 1; }
    }
    __syncthreads();

    // ---- parallel cluster-embedding merge ----
    if (decDo) {
      const int cid = decCid, ma = decMa, ini = decInit;
      const float size = decSize;
      const float inv = 1.0f / (size + 1.0f);
      float* crow = cemb + (size_t)cid * DSPAN;
      const float* arow = spanEmb + (size_t)ma * DSPAN;
      for (int k = tid; k < DSPAN; k += 1024) {
        float base = ini ? arow[k] : crow[k];     // init-then-merge, fused
        crow[k] = (base * size + span[k]) * inv;
      }
    }
    __threadfence_block();
    __syncthreads();
  }
}

extern "C" void kernel_launch(void* const* d_in, const int* in_sizes, int n_in,
                              void* d_out, int out_size, void* d_ws, size_t ws_size,
                              hipStream_t stream) {
  const float* spanEmb     = (const float*)d_in[0];
  const float* antScores   = (const float*)d_in[1];
  const float* clusSizeEmb = (const float*)d_in[2];
  const float* W1          = (const float*)d_in[3];
  const float* b1          = (const float*)d_in[4];
  const float* W2          = (const float*)d_in[5];
  const float* b2          = (const float*)d_in[6];
  const int*   antIdx      = (const int*)d_in[7];
  float* out = (float*)d_out;

  // ws layout: [Bpack: KPAD*512 bf16 = 7.0 MB][cemb: 2049*2324 f32 = 19.0 MB]
  __bf16* Bpack = (__bf16*)d_ws;
  float*  cemb  = (float*)((char*)d_ws + (size_t)KPAD * UN * sizeof(__bf16));

  {
    int total = KPAD * UN;
    pack_w1_bf16<<<(total + 255) / 256, 256, 0, stream>>>(W1, Bpack);
  }
  {
    int total = CROWS * DSPAN;
    zero_f32<<<(total + 255) / 256, 256, 0, stream>>>(cemb, total);
  }
  coref_main<<<1, 1024, 0, stream>>>(spanEmb, antScores, clusSizeEmb,
                                     b1, W2, b2, antIdx, Bpack, cemb, out);
}
